// KANLayer_47364899340538
// MI455X (gfx1250) — matrix operand order repaired and checked by reading, not compiled, not run on hardware
//
#include <hip/hip_runtime.h>

// ---------------------------------------------------------------------------
// KAN layer (cubic B-spline, uniform knots) as an LDS-staged WMMA GEMM.
//   y[b,o] = sum_{i,k} Bx[b,i,k] * (coeffs[o,i,k] * mask[o,i])
// Fast path (needs 12 MB of d_ws):
//   pass 0: pack coeffs*mask -> f16 (8 MB, L2-resident); transpose x -> xT.
//   pass 1: double-buffered WMMA GEMM; B tiles DMA'd to LDS with
//           global_load_async_to_lds_b128 (ASYNCcnt), A tiles (sparse cubic
//           spline rows) computed in-registers and scattered to LDS.
// ---------------------------------------------------------------------------

typedef __attribute__((ext_vector_type(16))) _Float16 v16h;
typedef __attribute__((ext_vector_type(8)))  float    v8f;

#define DIN   256
#define DOUT  256
#define KBAS  64      // N_KNOTS - P - 1
#define NKNOT 68
#define BM    128     // block tile rows (batch)
#define BN    64      // block tile cols (d_out)

union Frag { uint4 q[2]; v16h v; };

__device__ __forceinline__ void async_copy_b128(unsigned lds_addr,
                                                const void* gptr) {
    // GLOBAL_LOAD_ASYNC_TO_LDS_B128 (GV mode): per-lane 16B global -> LDS.
    asm volatile("global_load_async_to_lds_b128 %0, %1, off"
                 :: "v"(lds_addr), "v"(gptr) : "memory");
}
__device__ __forceinline__ void wait_async0() {
    asm volatile("s_wait_asynccnt 0x0" ::: "memory");
}

// ---------------------------------------------------------------------------
// Pack coeffs * mask into f16 (one float4 -> 4 halfs per thread).
// ---------------------------------------------------------------------------
__global__ __launch_bounds__(256)
void kan_pack_coeffs(const float* __restrict__ coeffs,
                     const float* __restrict__ mask,
                     _Float16* __restrict__ cpk)
{
    const size_t idx = (size_t)blockIdx.x * 256 + threadIdx.x;
    const size_t e   = idx * 4;
    const size_t oi  = e / KBAS;                   // o*DIN + i
    const float  mv  = mask[oi];
    const float4 c   = *(const float4*)(coeffs + e);
    _Float16* d = cpk + e;
    d[0] = (_Float16)(c.x * mv);
    d[1] = (_Float16)(c.y * mv);
    d[2] = (_Float16)(c.z * mv);
    d[3] = (_Float16)(c.w * mv);
}

// ---------------------------------------------------------------------------
// LDS-tiled transpose: x[b][i] -> xT[i][b]
// ---------------------------------------------------------------------------
__global__ __launch_bounds__(256)
void kan_transpose_x(const float* __restrict__ x,
                     float* __restrict__ xT, int Brows)
{
    __shared__ float tile[32][33];
    const int bx = blockIdx.x * 32;
    const int ix = blockIdx.y * 32;
    const int lx = threadIdx.x & 31;
    const int ly = threadIdx.x >> 5;
    #pragma unroll
    for (int r = ly; r < 32; r += 8)
        tile[r][lx] = x[(size_t)(bx + r) * DIN + ix + lx];
    __syncthreads();
    #pragma unroll
    for (int r = ly; r < 32; r += 8)
        xT[(size_t)(ix + r) * Brows + bx + lx] = tile[lx][r];
}

// ---------------------------------------------------------------------------
// Fast GEMM: double-buffered, async B-tile DMA, on-the-fly spline A tiles.
// ---------------------------------------------------------------------------
__global__ __launch_bounds__(256)
void kan_wmma_fast(const float* __restrict__ xT,
                   const _Float16* __restrict__ cpk,
                   const float* __restrict__ t,
                   float* __restrict__ out, int Brows)
{
    __shared__ __align__(16) _Float16 As[2][BM * KBAS];   // 32 KB [buf][row][k]
    __shared__ __align__(16) _Float16 Bs[2][BN * KBAS];   // 16 KB [buf][n][k]

    const int tid  = threadIdx.x;
    const int wave = tid >> 5;
    const int lane = tid & 31;
    const int half = lane >> 4;
    const int l16  = lane & 15;

    const int m0 = blockIdx.y * BM;
    const int n0 = blockIdx.x * BN;
    const int wm = wave >> 1;            // 4x2 wave grid over 128x64
    const int wn = wave & 1;

    const float t0   = t[0];
    const float hinv = 1.0f / (t[1] - t[0]);

    v8f acc[2][2];
    #pragma unroll
    for (int ms = 0; ms < 2; ++ms)
        #pragma unroll
        for (int ns = 0; ns < 2; ++ns)
            acc[ms][ns] = (v8f){0.f,0.f,0.f,0.f,0.f,0.f,0.f,0.f};

    // Stage tile i into LDS buffer `buf`.
    auto stage = [&](int i, int buf) {
        if (tid < BM) {
            // A tile: one sparse cubic-spline row per thread.
            const int r = tid;
            uint4* zp = (uint4*)&As[buf][r * KBAS];
            const uint4 z = {0u, 0u, 0u, 0u};
            #pragma unroll
            for (int q = 0; q < 8; ++q) zp[q] = z;

            const float u = xT[(size_t)i * Brows + m0 + r];
            if (i + 1 < DIN)
                __builtin_prefetch(&xT[(size_t)(i + 1) * Brows + m0 + r], 0, 1);

            const float pos = (u - t0) * hinv;
            const int   j   = (int)floorf(pos);
            if (j >= 0 && j < NKNOT - 1) {
                const float s  = pos - (float)j;
                const float os = 1.0f - s;
                const float s2 = s * s, s3 = s2 * s;
                float w[4];
                w[0] = os * os * os                        * (1.0f / 6.0f);
                w[1] = ( 3.0f*s3 - 6.0f*s2 + 4.0f)         * (1.0f / 6.0f);
                w[2] = (-3.0f*s3 + 3.0f*s2 + 3.0f*s + 1.0f)* (1.0f / 6.0f);
                w[3] = s3                                  * (1.0f / 6.0f);
                #pragma unroll
                for (int c = 0; c < 4; ++c) {
                    const int m = j - 3 + c;
                    if (m >= 0 && m < KBAS)
                        As[buf][r * KBAS + m] = (_Float16)w[c];
                }
            }
        } else {
            // B tile: 64 B per thread, DMA'd straight into LDS (ASYNCcnt).
            const int jx = tid - BM;
            const int ol = jx >> 1;
            const int kh = (jx & 1) * 32;
            const char* srcp = (const char*)
                (cpk + ((size_t)(n0 + ol) * DIN + i) * KBAS + kh);
            const unsigned lbase =
                (unsigned)(uintptr_t)&Bs[buf][ol * KBAS + kh];
            #pragma unroll
            for (int q = 0; q < 4; ++q)
                async_copy_b128(lbase + q * 16, srcp + q * 16);
        }
    };

    stage(0, 0);
    wait_async0();
    __syncthreads();

    for (int i = 0; i < DIN; ++i) {
        const int cur = i & 1;
        if (i + 1 < DIN) stage(i + 1, cur ^ 1);   // overlap with WMMA below

        #pragma unroll
        for (int kb = 0; kb < KBAS; kb += 32) {
            Frag a[2], b[2];
            #pragma unroll
            for (int ms = 0; ms < 2; ++ms) {
                const int row = wm * 32 + ms * 16 + l16;
                a[ms].q[0] = *(const uint4*)&As[cur][row * KBAS + kb + 8 * half];
                a[ms].q[1] = *(const uint4*)&As[cur][row * KBAS + kb + 16 + 8 * half];
            }
            #pragma unroll
            for (int ns = 0; ns < 2; ++ns) {
                const int col = wn * 32 + ns * 16 + l16;
                b[ns].q[0] = *(const uint4*)&Bs[cur][col * KBAS + kb + 8 * half];
                b[ns].q[1] = *(const uint4*)&Bs[cur][col * KBAS + kb + 16 + 8 * half];
            }
            #pragma unroll
            for (int ms = 0; ms < 2; ++ms)
                #pragma unroll
                for (int ns = 0; ns < 2; ++ns)
                    acc[ms][ns] = __builtin_amdgcn_wmma_f32_16x16x32_f16(
                        false, a[ms].v, false, b[ns].v,
                        (short)0, acc[ms][ns], false, false);
        }

        wait_async0();        // B tile of i+1 resident before barrier
        __syncthreads();      // one barrier per iteration (double-buffered)
    }

    // Epilogue: VGPR v -> M = v + 8*(lane>=16), N = lane&15.
    #pragma unroll
    for (int ms = 0; ms < 2; ++ms)
        #pragma unroll
        for (int ns = 0; ns < 2; ++ns) {
            const int col     = n0 + wn * 32 + ns * 16 + l16;
            const int rowbase = m0 + wm * 32 + ms * 16 + half * 8;
            #pragma unroll
            for (int v = 0; v < 8; ++v)
                out[(size_t)(rowbase + v) * DOUT + col] = acc[ms][ns][v];
        }
}

// ---------------------------------------------------------------------------
// Fallback (no workspace): f32 coeffs staged + masked in-loop.
// ---------------------------------------------------------------------------
__global__ __launch_bounds__(256)
void kan_wmma_f16(const float* __restrict__ x,
                  const float* __restrict__ coeffs,
                  const float* __restrict__ mask,
                  const float* __restrict__ t,
                  float* __restrict__ out)
{
    __shared__ __align__(16) _Float16 As[BM * KBAS];
    __shared__ __align__(16) _Float16 Bs[BN * KBAS];

    const int tid  = threadIdx.x;
    const int wave = tid >> 5;
    const int lane = tid & 31;
    const int half = lane >> 4;
    const int l16  = lane & 15;
    const int m0 = blockIdx.y * BM;
    const int n0 = blockIdx.x * BN;
    const int wm = wave >> 1;
    const int wn = wave & 1;

    const float t0   = t[0];
    const float hinv = 1.0f / (t[1] - t[0]);

    v8f acc[2][2];
    #pragma unroll
    for (int ms = 0; ms < 2; ++ms)
        #pragma unroll
        for (int ns = 0; ns < 2; ++ns)
            acc[ms][ns] = (v8f){0.f,0.f,0.f,0.f,0.f,0.f,0.f,0.f};

    for (int i = 0; i < DIN; ++i) {
        if (tid < BM) {
            const int r = tid;
            uint4* zp = (uint4*)&As[r * KBAS];
            const uint4 z = {0u, 0u, 0u, 0u};
            #pragma unroll
            for (int q = 0; q < 8; ++q) zp[q] = z;
            const float u   = x[(size_t)(m0 + r) * DIN + i];
            const float pos = (u - t0) * hinv;
            const int   j   = (int)floorf(pos);
            if (j >= 0 && j < NKNOT - 1) {
                const float s  = pos - (float)j;
                const float os = 1.0f - s;
                const float s2 = s * s, s3 = s2 * s;
                float w[4];
                w[0] = os * os * os                        * (1.0f / 6.0f);
                w[1] = ( 3.0f*s3 - 6.0f*s2 + 4.0f)         * (1.0f / 6.0f);
                w[2] = (-3.0f*s3 + 3.0f*s2 + 3.0f*s + 1.0f)* (1.0f / 6.0f);
                w[3] = s3                                  * (1.0f / 6.0f);
                #pragma unroll
                for (int c = 0; c < 4; ++c) {
                    const int m = j - 3 + c;
                    if (m >= 0 && m < KBAS)
                        As[r * KBAS + m] = (_Float16)w[c];
                }
            }
        } else {
            const int jx = tid - BM;
            const int ol = jx >> 1;
            const int kh = (jx & 1) * 32;
            const int o  = n0 + ol;
            const float mv = mask[(size_t)o * DIN + i];
            const float4* src =
                (const float4*)&coeffs[((size_t)o * DIN + i) * KBAS + kh];
            #pragma unroll
            for (int q = 0; q < 8; ++q) {
                const float4 c4 = src[q];
                _Float16* dst = &Bs[ol * KBAS + kh + q * 4];
                dst[0] = (_Float16)(c4.x * mv);
                dst[1] = (_Float16)(c4.y * mv);
                dst[2] = (_Float16)(c4.z * mv);
                dst[3] = (_Float16)(c4.w * mv);
            }
        }
        __syncthreads();

        #pragma unroll
        for (int kb = 0; kb < KBAS; kb += 32) {
            Frag a[2], b[2];
            #pragma unroll
            for (int ms = 0; ms < 2; ++ms) {
                const int row = wm * 32 + ms * 16 + l16;
                a[ms].q[0] = *(const uint4*)&As[row * KBAS + kb + 8 * half];
                a[ms].q[1] = *(const uint4*)&As[row * KBAS + kb + 16 + 8 * half];
            }
            #pragma unroll
            for (int ns = 0; ns < 2; ++ns) {
                const int col = wn * 32 + ns * 16 + l16;
                b[ns].q[0] = *(const uint4*)&Bs[col * KBAS + kb + 8 * half];
                b[ns].q[1] = *(const uint4*)&Bs[col * KBAS + kb + 16 + 8 * half];
            }
            #pragma unroll
            for (int ms = 0; ms < 2; ++ms)
                #pragma unroll
                for (int ns = 0; ns < 2; ++ns)
                    acc[ms][ns] = __builtin_amdgcn_wmma_f32_16x16x32_f16(
                        false, a[ms].v, false, b[ns].v,
                        (short)0, acc[ms][ns], false, false);
        }
        __syncthreads();
    }

    #pragma unroll
    for (int ms = 0; ms < 2; ++ms)
        #pragma unroll
        for (int ns = 0; ns < 2; ++ns) {
            const int col     = n0 + wn * 32 + ns * 16 + l16;
            const int rowbase = m0 + wm * 32 + ms * 16 + half * 8;
            #pragma unroll
            for (int v = 0; v < 8; ++v)
                out[(size_t)(rowbase + v) * DOUT + col] = acc[ms][ns][v];
        }
}

extern "C" void kernel_launch(void* const* d_in, const int* in_sizes, int n_in,
                              void* d_out, int out_size, void* d_ws, size_t ws_size,
                              hipStream_t stream) {
    const float* x      = (const float*)d_in[0];   // (4096, 256)
    const float* coeffs = (const float*)d_in[1];   // (256, 256, 64)
    const float* mask   = (const float*)d_in[2];   // (256, 256)
    const float* t      = (const float*)d_in[3];   // (68,)
    float* out = (float*)d_out;                    // (4096, 256)

    const int Brows = in_sizes[0] / DIN;           // 4096
    const size_t cpk_bytes = (size_t)DOUT * DIN * KBAS * sizeof(_Float16);
    const size_t xT_bytes  = (size_t)Brows * DIN * sizeof(float);

    dim3 grid(DOUT / BN, Brows / BM);              // (4, 32)

    if (ws_size >= cpk_bytes + xT_bytes) {
        _Float16* cpk = (_Float16*)d_ws;
        float*    xT  = (float*)((char*)d_ws + cpk_bytes);
        const int nblk = (DOUT * DIN * KBAS / 4 + 255) / 256;   // 4096
        kan_pack_coeffs<<<nblk, 256, 0, stream>>>(coeffs, mask, cpk);
        dim3 tg(Brows / 32, DIN / 32);                          // (128, 8)
        kan_transpose_x<<<tg, 256, 0, stream>>>(x, xT, Brows);
        kan_wmma_fast<<<grid, 256, 0, stream>>>(xT, cpk, t, out, Brows);
    } else {
        kan_wmma_f16<<<grid, 256, 0, stream>>>(x, coeffs, mask, t, out);
    }
}
